// DPF_68023692034282
// MI455X (gfx1250) — compile-verified, matrix operand order checked by reading, not run on hardware
//
#include <hip/hip_runtime.h>
#include <hip/hip_bf16.h>
#include <stddef.h>
#include <stdint.h>

// ---------------------------------------------------------------------------
// MI455X (gfx1250, wave32) fused complement-attention block.
//
// Roofline: ~58 GFLOP, ~48 MB essential traffic. Materializing the 4x4096x4096
// attention matrix would add ~1 GB of HBM traffic (~45us @ 23.3 TB/s), so we
// fuse the softmax/AV stage flash-attention style and use the identity
//   (1 - softmax(S)) @ V = colsum(V) - softmax(S) @ V.
// All matmuls use v_wmma_f32_16x16x32_f16 (f16 in, f32 accumulate).
//
// Attention inner loop double-buffers the K/V tiles in LDS using the CDNA5
// async engine (global_load_async_to_lds_b128 + s_wait_asynccnt) so HBM
// latency for tile m0+32 overlaps the 24 WMMAs of tile m0, without burning
// VGPRs on load-to-use distance (the kernel is already >256 VGPRs).
// ---------------------------------------------------------------------------

typedef __attribute__((ext_vector_type(16))) _Float16 v16h;
typedef __attribute__((ext_vector_type(8)))  _Float16 v8h;
typedef __attribute__((ext_vector_type(8)))  float    v8f;

#define WMMA_F16(A, B, C) \
  __builtin_amdgcn_wmma_f32_16x16x32_f16(false, (A), false, (B), (short)0, (C), false, false)

// 16-bit A/B fragment layout (ISA 7.12.2): lane l holds row/col (l&15),
// kbase = (l>>4)*8, half h maps to K = kbase + (h>=8)*16 + (h&7).
// => two contiguous 8-half (16B) segments at +0 and +16 from (base+kbase).
__device__ __forceinline__ v16h load_frag_h(const _Float16* base) {
  v8h lo = *(const v8h*)(base);
  v8h hi = *(const v8h*)(base + 16);
  v16h r;
#pragma unroll
  for (int i = 0; i < 8; ++i) { r[i] = lo[i]; r[8 + i] = hi[i]; }
  return r;
}

__device__ __forceinline__ v16h load_frag_f32(const float* base) {
  v16h r;
#pragma unroll
  for (int i = 0; i < 8; ++i) {
    r[i]     = (_Float16)base[i];
    r[8 + i] = (_Float16)base[16 + i];
  }
  return r;
}

// Same two-segment fragment gather, but from LDS (emits ds_load_b128 pairs).
__device__ __forceinline__ v16h frag_from_lds(const _Float16* base) {
  v16h r;
#pragma unroll
  for (int i = 0; i < 8; ++i) { r[i] = base[i]; r[8 + i] = base[16 + i]; }
  return r;
}

// CDNA5 async copy: per lane LDS[vdst] = MEM[vaddr..+15]; tracked by ASYNCcnt.
// (ISA §15.18.3 opcode 98, GV addressing.) Generic->LDS: low 32 bits of a
// __shared__ generic address are the LDS byte offset.
__device__ __forceinline__ void async_b128(const _Float16* lds_dst,
                                           const _Float16* gsrc) {
  uint32_t ldsoff = (uint32_t)(uintptr_t)lds_dst;
  uint64_t gaddr  = (uint64_t)(uintptr_t)gsrc;
  asm volatile("global_load_async_to_lds_b128 %0, %1, off"
               :: "v"(ldsoff), "v"(gaddr) : "memory");
}

// ---------------------------------------------------------------------------
// Projection: y = relu((W @ x + b) * s + t), x is [C, Ntot] (channels-major).
// GEMM orientation: M = out-channel tile (A = W rows, contiguous K reads),
// N = pixel tile (B = x tile, per-K coalesced row reads), K = C.
// TRANS_OUT: true  -> y stored [n, OC] (row-major per pixel; for Q, K)
//            false -> y stored [OC, Ntot] (channels-major; for V)
// ---------------------------------------------------------------------------
template <bool TRANS_OUT>
__global__ void __launch_bounds__(32)
proj_kernel(const float* __restrict__ x, const float* __restrict__ W,
            const float* __restrict__ bias, const float* __restrict__ sc,
            const float* __restrict__ tr, _Float16* __restrict__ y,
            int C, int OC, int Ntot) {
  const int l  = threadIdx.x;
  const int lo = l & 15;
  const int hb = l >> 4;
  const int kb = hb * 8;
  const int n0 = blockIdx.x * 16;
  const int o0 = blockIdx.y * 16;
  const int b  = blockIdx.z;

  const float* xb = x + (size_t)b * C * Ntot;
  _Float16*    yb = y + (size_t)b * OC * Ntot;

  v8f acc = {};
  for (int c0 = 0; c0 < C; c0 += 32) {
    // A: W tile (16 o x 32 c); lane row o = o0+lo, contiguous c.
    v16h afr = load_frag_f32(W + (size_t)(o0 + lo) * C + c0 + kb);
    // B: x tile (32 c x 16 n); lane col n = n0+lo, strided c (coalesced per K).
    v16h bfr;
#pragma unroll
    for (int h = 0; h < 16; ++h) {
      int c = c0 + kb + ((h >= 8) ? 16 : 0) + (h & 7);
      bfr[h] = (_Float16)xb[(size_t)c * Ntot + n0 + lo];
    }
    acc = WMMA_F16(afr, bfr, acc);
  }

#pragma unroll
  for (int v = 0; v < 8; ++v) {
    int o = o0 + hb * 8 + v;   // D layout: M = v + (l>>4)*8
    int n = n0 + lo;           //           N = l & 15
    float r = (acc[v] + bias[o]) * sc[o] + tr[o];
    r = fmaxf(r, 0.0f);
    if (TRANS_OUT) yb[(size_t)n * OC + o] = (_Float16)r;
    else           yb[(size_t)o * Ntot + n] = (_Float16)r;
  }
}

// ---------------------------------------------------------------------------
// Column-sum of V over pixels: csum[b, c] = sum_m V[b, c, m]   (V is [c, Ntot])
// ---------------------------------------------------------------------------
__global__ void __launch_bounds__(256)
colsum_kernel(const _Float16* __restrict__ V, float* __restrict__ csum, int Ntot) {
  const int c = blockIdx.x, b = blockIdx.y, tid = threadIdx.x;
  const _Float16* row = V + ((size_t)b * 128 + c) * Ntot;
  float s = 0.0f;
  for (int i = tid; i < Ntot; i += 256) s += (float)row[i];
  __shared__ float red[256];
  red[tid] = s;
  __syncthreads();
#pragma unroll
  for (int off = 128; off > 0; off >>= 1) {
    if (tid < off) red[tid] += red[tid + off];
    __syncthreads();
  }
  if (tid == 0) csum[(size_t)b * 128 + c] = red[0];
}

// ---------------------------------------------------------------------------
// Fused flash-attention with complement. One wave owns 16 query rows; streams
// key/value columns 32 at a time. K/V tiles are double-buffered in LDS via
// async loads (16 + 16 + 16 b128 ops = 48 per tile, 512 B each); we wait with
// s_wait_asynccnt 48 so the next tile's loads stay in flight during compute.
// Online softmax stats per row; P converted to an A-fragment through LDS.
// Epilogue: out = colsum(V) - O/l, written as f16 into cat.
// ---------------------------------------------------------------------------
__global__ void __launch_bounds__(32)
attn_kernel(const _Float16* __restrict__ Qh, const _Float16* __restrict__ Kh,
            const _Float16* __restrict__ V1h, const _Float16* __restrict__ V2h,
            const float* __restrict__ csum1, const float* __restrict__ csum2,
            _Float16* __restrict__ cat, int Ntot) {
  const int l  = threadIdx.x;
  const int lo = l & 15;
  const int hb = l >> 4;
  const int kb = hb * 8;
  const int n0 = blockIdx.x * 16;
  const int b  = blockIdx.y;

  const _Float16* Q  = Qh  + (size_t)b * Ntot * 128;
  const _Float16* K2 = Kh  + (size_t)b * Ntot * 128;
  const _Float16* V1 = V1h + (size_t)b * 128 * Ntot;
  const _Float16* V2 = V2h + (size_t)b * 128 * Ntot;
  const float*    c1 = csum1 + (size_t)b * 128;
  const float*    c2 = csum2 + (size_t)b * 128;
  _Float16*       Ct = cat + (size_t)b * Ntot * 256;

  // Double-buffered tiles: K rows [32][128], V tiles [128][32] each. 48 KB.
  __shared__ __align__(16) _Float16 kbuf[2][32 * 128];
  __shared__ __align__(16) _Float16 v1buf[2][128 * 32];
  __shared__ __align__(16) _Float16 v2buf[2][128 * 32];
  __shared__ __align__(16) _Float16 ptile[16 * 32];

  // Per-lane async-copy geometry.
  const int krow = hb;            // K: 2 rows (256 B each) per b128 op
  const int kcol = lo * 8;        //    16 B per lane along c
  const int vrow = l >> 2;        // V: 8 rows (64 B each) per b128 op
  const int vcol = (l & 3) * 8;   //    16 B per lane along m

  // Issue one stage: 48 async b128 ops (K tile + V1 tile + V2 tile).
  auto issue_tile = [&](int m0, int s) {
#pragma unroll
    for (int i = 0; i < 16; ++i) {
      int r = 2 * i + krow;
      async_b128(&kbuf[s][r * 128 + kcol],
                 K2 + (size_t)(m0 + r) * 128 + kcol);
    }
#pragma unroll
    for (int i = 0; i < 16; ++i) {
      int c = 8 * i + vrow;
      async_b128(&v1buf[s][c * 32 + vcol], V1 + (size_t)c * Ntot + m0 + vcol);
    }
#pragma unroll
    for (int i = 0; i < 16; ++i) {
      int c = 8 * i + vrow;
      async_b128(&v2buf[s][c * 32 + vcol], V2 + (size_t)c * Ntot + m0 + vcol);
    }
  };

  // Resident Q fragments: 16 rows x 128 ch = 4 fragments of K=32.
  v16h qa[4];
#pragma unroll
  for (int f = 0; f < 4; ++f)
    qa[f] = load_frag_h(Q + (size_t)(n0 + lo) * 128 + f * 32 + kb);

  v8f o1[8], o2[8];
#pragma unroll
  for (int t = 0; t < 8; ++t) { o1[t] = (v8f){}; o2[t] = (v8f){}; }
  float m_run[8], l_run[8];
#pragma unroll
  for (int v = 0; v < 8; ++v) { m_run[v] = -1e30f; l_run[v] = 0.0f; }

  const float scale = 0.08838834764831845f;  // 1/sqrt(128)

  issue_tile(0, 0);

  for (int m0 = 0; m0 < Ntot; m0 += 32) {
    const int s = (m0 >> 5) & 1;
    if (m0 + 32 < Ntot) {
      issue_tile(m0 + 32, s ^ 1);
      // Current stage (oldest 48) done; next stage's 48 remain in flight.
      asm volatile("s_wait_asynccnt 48" ::: "memory");
    } else {
      asm volatile("s_wait_asynccnt 0" ::: "memory");
    }

    // S tiles: 16 rows x 32 cols, K = 128 in 4 steps; K fragments from LDS.
    v8f s0 = {}, s1 = {};
#pragma unroll
    for (int f = 0; f < 4; ++f) {
      v16h kf0 = frag_from_lds(&kbuf[s][lo * 128 + f * 32 + kb]);
      v16h kf1 = frag_from_lds(&kbuf[s][(16 + lo) * 128 + f * 32 + kb]);
      s0 = WMMA_F16(qa[f], kf0, s0);
      s1 = WMMA_F16(qa[f], kf1, s1);
    }

    // Online softmax across the 32 new columns (row = v + hb*8; its 16 column
    // entries live across the 16 lanes of this half-wave).
#pragma unroll
    for (int v = 0; v < 8; ++v) {
      float a = s0[v] * scale;
      float c = s1[v] * scale;
      float mx = fmaxf(a, c);
#pragma unroll
      for (int off = 8; off >= 1; off >>= 1)
        mx = fmaxf(mx, __shfl_xor(mx, off, 16));
      float mnew = fmaxf(m_run[v], mx);
      float corr = __expf(m_run[v] - mnew);
      float p0 = __expf(a - mnew);
      float p1 = __expf(c - mnew);
      float ls = p0 + p1;
#pragma unroll
      for (int off = 8; off >= 1; off >>= 1)
        ls += __shfl_xor(ls, off, 16);
      l_run[v] = l_run[v] * corr + ls;
      m_run[v] = mnew;
#pragma unroll
      for (int t = 0; t < 8; ++t) { o1[t][v] *= corr; o2[t][v] *= corr; }
      // Stash P (C/D layout) into LDS for re-read in A layout.
      ptile[(hb * 8 + v) * 32 + lo]      = (_Float16)p0;
      ptile[(hb * 8 + v) * 32 + 16 + lo] = (_Float16)p1;
    }
    __syncthreads();

    // Re-read P as a 16x32 A-fragment.
    v16h pa;
#pragma unroll
    for (int h = 0; h < 16; ++h)
      pa[h] = ptile[lo * 32 + kb + ((h >= 8) ? 16 : 0) + (h & 7)];
    __syncthreads();

    // O += P @ V for both value streams; V tiles in LDS as [c][m] so the
    // B-fragment reads (lane col = out channel, contiguous K = m) are
    // contiguous ds_load_b128 pairs.
#pragma unroll
    for (int t = 0; t < 8; ++t) {
      v16h vb2 = frag_from_lds(&v2buf[s][(t * 16 + lo) * 32 + kb]);
      v16h vb1 = frag_from_lds(&v1buf[s][(t * 16 + lo) * 32 + kb]);
      o1[t] = WMMA_F16(pa, vb2, o1[t]);
      o2[t] = WMMA_F16(pa, vb1, o2[t]);
    }
  }

  // Epilogue: complement + concat. out1 = colsum(V2) - softmax@V2 (uses V2),
  // out2 = colsum(V1) - softmax@V1. cat layout: [n, 256], out1 at 0..127.
#pragma unroll
  for (int t = 0; t < 8; ++t) {
#pragma unroll
    for (int v = 0; v < 8; ++v) {
      int og = t * 16 + lo;
      int n  = n0 + hb * 8 + v;
      float inv = 1.0f / l_run[v];
      float r1 = c2[og] - o1[t][v] * inv;
      float r2 = c1[og] - o2[t][v] * inv;
      Ct[(size_t)n * 256 + og]       = (_Float16)r1;
      Ct[(size_t)n * 256 + 128 + og] = (_Float16)r2;
    }
  }
}

// ---------------------------------------------------------------------------
// Output projection: out[b,o,n] = relu((Wp @ cat[n,:] + bp)*sp + tp), f32 out.
// M = out-channel, N = pixel => coalesced 32-bit stores along n.
// ---------------------------------------------------------------------------
__global__ void __launch_bounds__(32)
outproj_kernel(const _Float16* __restrict__ cat, const float* __restrict__ Wp,
               const float* __restrict__ bp, const float* __restrict__ sp,
               const float* __restrict__ tp, float* __restrict__ out, int Ntot) {
  const int l  = threadIdx.x;
  const int lo = l & 15;
  const int hb = l >> 4;
  const int kb = hb * 8;
  const int n0 = blockIdx.x * 16;
  const int o0 = blockIdx.y * 16;
  const int b  = blockIdx.z;

  const _Float16* Cb = cat + (size_t)b * Ntot * 256;
  float*          Ob = out + (size_t)b * 256 * Ntot;

  v8f acc = {};
  for (int k0 = 0; k0 < 256; k0 += 32) {
    v16h afr = load_frag_f32(Wp + (size_t)(o0 + lo) * 256 + k0 + kb);
    v16h bfr = load_frag_h(Cb + (size_t)(n0 + lo) * 256 + k0 + kb);
    acc = WMMA_F16(afr, bfr, acc);
  }

#pragma unroll
  for (int v = 0; v < 8; ++v) {
    int o = o0 + hb * 8 + v;
    int n = n0 + lo;
    float r = (acc[v] + bp[o]) * sp[o] + tp[o];
    Ob[(size_t)o * Ntot + n] = fmaxf(r, 0.0f);
  }
}

// ---------------------------------------------------------------------------
extern "C" void kernel_launch(void* const* d_in, const int* in_sizes, int n_in,
                              void* d_out, int out_size, void* d_ws, size_t ws_size,
                              hipStream_t stream) {
  const int B = 4, C = 256, Ntot = 64 * 64, INTER = 128, OUT = 256;

  const float* x1 = (const float*)d_in[0];
  const float* x2 = (const float*)d_in[1];
  const float* Wq = (const float*)d_in[2];
  const float* bq = (const float*)d_in[3];
  const float* sq = (const float*)d_in[4];
  const float* tq = (const float*)d_in[5];
  const float* Wk = (const float*)d_in[6];
  const float* bk = (const float*)d_in[7];
  const float* sk = (const float*)d_in[8];
  const float* tk = (const float*)d_in[9];
  const float* Wv = (const float*)d_in[10];
  const float* bv = (const float*)d_in[11];
  const float* sv = (const float*)d_in[12];
  const float* tv = (const float*)d_in[13];
  const float* Wp = (const float*)d_in[14];
  const float* bp = (const float*)d_in[15];
  const float* sp = (const float*)d_in[16];
  const float* tp = (const float*)d_in[17];
  float* out = (float*)d_out;

  // Workspace layout (bytes).
  char* ws = (char*)d_ws;
  const size_t szProj = (size_t)B * Ntot * INTER * sizeof(_Float16);  // 4 MB
  _Float16* Qh  = (_Float16*)(ws);                       // [b, n, 128]
  _Float16* Kh  = (_Float16*)(ws + szProj);              // [b, m, 128]
  _Float16* V1h = (_Float16*)(ws + 2 * szProj);          // [b, 128, n]
  _Float16* V2h = (_Float16*)(ws + 3 * szProj);          // [b, 128, n]
  _Float16* cat = (_Float16*)(ws + 4 * szProj);          // [b, n, 256], 8 MB
  float* csum1  = (float*)(ws + 6 * szProj);
  float* csum2  = csum1 + (size_t)B * INTER;

  dim3 blk(32, 1, 1);
  dim3 gProj(Ntot / 16, INTER / 16, B);
  proj_kernel<true ><<<gProj, blk, 0, stream>>>(x1, Wq, bq, sq, tq, Qh,  C, INTER, Ntot);
  proj_kernel<true ><<<gProj, blk, 0, stream>>>(x2, Wk, bk, sk, tk, Kh,  C, INTER, Ntot);
  proj_kernel<false><<<gProj, blk, 0, stream>>>(x1, Wv, bv, sv, tv, V1h, C, INTER, Ntot);
  proj_kernel<false><<<gProj, blk, 0, stream>>>(x2, Wv, bv, sv, tv, V2h, C, INTER, Ntot);

  dim3 gSum(INTER, B);
  colsum_kernel<<<gSum, dim3(256), 0, stream>>>(V1h, csum1, Ntot);
  colsum_kernel<<<gSum, dim3(256), 0, stream>>>(V2h, csum2, Ntot);

  dim3 gAttn(Ntot / 16, B);
  attn_kernel<<<gAttn, blk, 0, stream>>>(Qh, Kh, V1h, V2h, csum1, csum2, cat, Ntot);

  dim3 gOut(Ntot / 16, OUT / 16, B);
  outproj_kernel<<<gOut, blk, 0, stream>>>(cat, Wp, bp, sp, tp, out, Ntot);

  (void)in_sizes; (void)n_in; (void)out_size; (void)ws_size;
}
